// MessagePassingLayer_13709535609413
// MI455X (gfx1250) — matrix operand order repaired and checked by reading, not compiled, not run on hardware
//
#include <hip/hip_runtime.h>
#include <hip/hip_bf16.h>

// ---------------------------------------------------------------------------
// MessagePassingLayer for MI455X (gfx1250, wave32, WMMA bf16 16x16x32)
//   N=100000 nodes, E=1600000 edges, H=64, EF=16
//   edge MLP (144->64 relu ->64) -> atomic segment-sum -> node MLP (128->64
//   relu ->64) -> residual + LayerNorm.
// GEMMs run on v_wmma_f32_16x16x32_bf16 (bf16 operands, fp32 accumulate).
// Node features are pre-converted to bf16 in d_ws so per-edge gathers are
// two 16B loads per A-fragment with no conversion VALU in the hot loop.
// ---------------------------------------------------------------------------

typedef __attribute__((ext_vector_type(16))) __bf16 v16bf;
typedef __attribute__((ext_vector_type(8)))  float  v8f;

struct U8 { uint4 a, b; };   // 32 bytes == v16bf

// fp32 pair -> packed bf16 (RNE). Prefer the native packed convert.
__device__ inline unsigned packbf(float lo, float hi) {
#if __has_builtin(__builtin_amdgcn_cvt_pk_bf16_f32)
    auto p = __builtin_amdgcn_cvt_pk_bf16_f32(lo, hi);
    return __builtin_bit_cast(unsigned, p);
#else
    struct { __bf16 x, y; } p = { (__bf16)lo, (__bf16)hi };
    return __builtin_bit_cast(unsigned, p);
#endif
}

__device__ inline v8f wmma_bf16(v16bf a, v16bf b, v8f c) {
    return __builtin_amdgcn_wmma_f32_16x16x32_bf16(
        false, a, false, b, (short)0, c, false, false);
}

// Build a 16x32 bf16 A fragment from 16 fp32 values (ISA 7.12.2 A layout:
// lanes 0-15 hold K = half*8..+7 in VGPR0-3 and K = 16+half*8..+7 in VGPR4-7).
__device__ inline v16bf makeA(float4 x0, float4 x1, float4 y0, float4 y1) {
    U8 u;
    u.a.x = packbf(x0.x, x0.y);  u.a.y = packbf(x0.z, x0.w);
    u.a.z = packbf(x1.x, x1.y);  u.a.w = packbf(x1.z, x1.w);
    u.b.x = packbf(y0.x, y0.y);  u.b.y = packbf(y0.z, y0.w);
    u.b.z = packbf(y1.x, y1.y);  u.b.w = packbf(y1.z, y1.w);
    return __builtin_bit_cast(v16bf, u);
}

// A fragment straight from a bf16 row: two 16B loads, zero conversion.
// row points at feature-chunk base (32 bf16), o = half*8.
__device__ inline v16bf ldAbf(const unsigned short* row, int o) {
    U8 u;
    u.a = *(const uint4*)(row + o);        // K = o .. o+7
    u.b = *(const uint4*)(row + 16 + o);   // K = 16+o .. 16+o+7
    return __builtin_bit_cast(v16bf, u);
}

// Load one 32x16 bf16 B-tile fragment (pre-staged in LDS, lane-major layout).
__device__ inline v16bf loadB(const unsigned* tileBase, int lane) {
    const unsigned* p = tileBase + lane * 8;
    U8 u;
    u.a = *(const uint4*)(p);
    u.b = *(const uint4*)(p + 4);
    return __builtin_bit_cast(v16bf, u);
}

#define HID_STRIDE 68   // 64 + 4 pad floats -> kills LDS bank conflicts

// ---------------------------------------------------------------------------
// Kernel: zero the aggregation workspace (N*64 floats)
// ---------------------------------------------------------------------------
__global__ void zero_f4_kernel(float4* __restrict__ p, int n4) {
    for (int i = blockIdx.x * blockDim.x + threadIdx.x; i < n4;
         i += gridDim.x * blockDim.x)
        p[i] = make_float4(0.f, 0.f, 0.f, 0.f);
}

// ---------------------------------------------------------------------------
// Kernel: fp32 -> packed bf16 (node-feature copy), npairs = count/2
// ---------------------------------------------------------------------------
__global__ void f32_to_bf16_kernel(const float* __restrict__ in,
                                   unsigned* __restrict__ out, int npairs) {
    for (int i = blockIdx.x * blockDim.x + threadIdx.x; i < npairs;
         i += gridDim.x * blockDim.x) {
        float2 v = *(const float2*)(in + 2 * (long)i);
        out[i] = packbf(v.x, v.y);
    }
}

// ---------------------------------------------------------------------------
// Kernel 1: edge MLP + atomic scatter-add into agg
//   BF16G=true : gather node rows from the pre-converted bf16 copy
//   BF16G=false: gather fp32 rows and convert in-loop (ws too small)
// ---------------------------------------------------------------------------
template <bool BF16G>
__global__ __launch_bounds__(256)
void edge_mlp_kernel(const float* __restrict__ nodef,
                     const unsigned short* __restrict__ nodefbf,
                     const long long* __restrict__ eidx,   // [2][E] row0=src row1=dst
                     const float* __restrict__ efeat,
                     const float* __restrict__ Wm1, const float* __restrict__ bm1,
                     const float* __restrict__ Wm2, const float* __restrict__ bm2,
                     float* __restrict__ agg,
                     int E, int ntiles) {
    __shared__ __align__(16) unsigned sW1[20 * 256];       // 5 kchunks x 4 ntiles
    __shared__ __align__(16) unsigned sW2[8 * 256];        // 2 kchunks x 4 ntiles
    __shared__ __align__(16) float    sHid[8][16 * HID_STRIDE];

    const int tid = threadIdx.x;

    // Stage W_m1 (144x64, rows 144..159 zero-padded) as bf16 B tiles
    for (int i = tid; i < 20 * 256; i += 256) {
        int tile = i >> 8;            // kc*4 + nt
        int lane = (i >> 3) & 31;
        int v    = i & 7;
        int kc = tile >> 2, nt = tile & 3;
        int half = lane >> 4, n = lane & 15;
        int k0 = kc * 32 + half * 8 + 2 * v + (v >= 4 ? 8 : 0);
        int col = nt * 16 + n;
        float lo = (k0     < 144) ? Wm1[k0 * 64 + col]       : 0.f;
        float hi = (k0 + 1 < 144) ? Wm1[(k0 + 1) * 64 + col] : 0.f;
        sW1[i] = packbf(lo, hi);
    }
    for (int i = tid; i < 8 * 256; i += 256) {
        int tile = i >> 8;
        int lane = (i >> 3) & 31;
        int v    = i & 7;
        int kc = tile >> 2, nt = tile & 3;
        int half = lane >> 4, n = lane & 15;
        int k0 = kc * 32 + half * 8 + 2 * v + (v >= 4 ? 8 : 0);
        int col = nt * 16 + n;
        sW2[i] = packbf(Wm2[k0 * 64 + col], Wm2[(k0 + 1) * 64 + col]);
    }
    __syncthreads();

    const int lane = tid & 31;
    const int wave = tid >> 5;
    const int half = lane >> 4;
    const int m    = lane & 15;
    const int n16  = lane & 15;
    float* hid = &sHid[wave][0];

    float bias1[4], bias2[4];
#pragma unroll
    for (int nt = 0; nt < 4; ++nt) {
        bias1[nt] = bm1[nt * 16 + n16];
        bias2[nt] = bm2[nt * 16 + n16];
    }

    const long long* srcRow = eidx;
    const long long* dstRow = eidx + E;

    const int gwave  = blockIdx.x * 8 + wave;
    const int nwaves = gridDim.x * 8;
    const int o = half * 8;

    for (int t = gwave; t < ntiles; t += nwaves) {
        const int e0  = t * 16;
        const int e_m = min(e0 + m, E - 1);
        const int sI  = (int)srcRow[e_m];
        const int dI  = (int)dstRow[e_m];

        v16bf Ad0, Ad1, As0, As1;
        if constexpr (BF16G) {
            const unsigned short* hb = nodefbf + (long)dI * 64;
            const unsigned short* sb = nodefbf + (long)sI * 64;
            Ad0 = ldAbf(hb,      o);
            Ad1 = ldAbf(hb + 32, o);
            As0 = ldAbf(sb,      o);
            As1 = ldAbf(sb + 32, o);
        } else {
            const float* hd = nodef + (long)dI * 64;
            const float* hs = nodef + (long)sI * 64;
            Ad0 = makeA(*(const float4*)(hd + o),      *(const float4*)(hd + o + 4),
                        *(const float4*)(hd + 16 + o), *(const float4*)(hd + 20 + o));
            Ad1 = makeA(*(const float4*)(hd + 32 + o), *(const float4*)(hd + 36 + o),
                        *(const float4*)(hd + 48 + o), *(const float4*)(hd + 52 + o));
            As0 = makeA(*(const float4*)(hs + o),      *(const float4*)(hs + o + 4),
                        *(const float4*)(hs + 16 + o), *(const float4*)(hs + 20 + o));
            As1 = makeA(*(const float4*)(hs + 32 + o), *(const float4*)(hs + 36 + o),
                        *(const float4*)(hs + 48 + o), *(const float4*)(hs + 52 + o));
        }
        const float* ep = efeat + (long)e_m * 16;
        const float4 z4 = make_float4(0.f, 0.f, 0.f, 0.f);
        v16bf Ae = makeA(*(const float4*)(ep + o), *(const float4*)(ep + o + 4), z4, z4);

        // ---- layer 1: (16x144) @ (144x64), relu, stage rows into LDS ----
#pragma unroll
        for (int nt = 0; nt < 4; ++nt) {
            float b = bias1[nt];
            v8f c = {b, b, b, b, b, b, b, b};
            c = wmma_bf16(Ad0, loadB(&sW1[(0 * 4 + nt) * 256], lane), c);
            c = wmma_bf16(Ad1, loadB(&sW1[(1 * 4 + nt) * 256], lane), c);
            c = wmma_bf16(As0, loadB(&sW1[(2 * 4 + nt) * 256], lane), c);
            c = wmma_bf16(As1, loadB(&sW1[(3 * 4 + nt) * 256], lane), c);
            c = wmma_bf16(Ae,  loadB(&sW1[(4 * 4 + nt) * 256], lane), c);
#pragma unroll
            for (int r = 0; r < 8; ++r)
                hid[(r + 8 * half) * HID_STRIDE + nt * 16 + n16] = fmaxf(c[r], 0.f);
        }

        // ---- re-tile hidden (row m) as layer-2 A fragments ----
        v16bf A2[2];
#pragma unroll
        for (int kc = 0; kc < 2; ++kc) {
            const float* hrow = hid + m * HID_STRIDE + kc * 32;
            A2[kc] = makeA(*(const float4*)(hrow + o),      *(const float4*)(hrow + o + 4),
                           *(const float4*)(hrow + 16 + o), *(const float4*)(hrow + 20 + o));
        }

        // ---- layer 2 + scatter-add (segment_sum over dst) ----
#pragma unroll
        for (int nt = 0; nt < 4; ++nt) {
            float b = bias2[nt];
            v8f c = {b, b, b, b, b, b, b, b};
            c = wmma_bf16(A2[0], loadB(&sW2[(0 * 4 + nt) * 256], lane), c);
            c = wmma_bf16(A2[1], loadB(&sW2[(1 * 4 + nt) * 256], lane), c);
#pragma unroll
            for (int r = 0; r < 8; ++r) {
                int mm = r + 8 * half;
                int di = __shfl(dI, mm, 32);
                if (e0 + mm < E)
                    atomicAdd(&agg[(long)di * 64 + nt * 16 + n16], c[r]);
            }
        }
    }
}

// ---------------------------------------------------------------------------
// Kernel 3: node update MLP + residual + LayerNorm
// ---------------------------------------------------------------------------
__global__ __launch_bounds__(256)
void node_update_kernel(const float* __restrict__ nodef,
                        const float* __restrict__ agg,
                        const float* __restrict__ Wu1, const float* __restrict__ bu1,
                        const float* __restrict__ Wu2, const float* __restrict__ bu2,
                        const float* __restrict__ gamma, const float* __restrict__ beta,
                        float* __restrict__ out,
                        int N, int ntiles) {
    __shared__ __align__(16) unsigned sW1[16 * 256];   // 4 kchunks x 4 ntiles
    __shared__ __align__(16) unsigned sW2[8 * 256];    // 2 kchunks x 4 ntiles
    __shared__ __align__(16) float    sHid[8][16 * HID_STRIDE];

    const int tid = threadIdx.x;

    for (int i = tid; i < 16 * 256; i += 256) {
        int tile = i >> 8;
        int lane = (i >> 3) & 31;
        int v    = i & 7;
        int kc = tile >> 2, nt = tile & 3;
        int half = lane >> 4, n = lane & 15;
        int k0 = kc * 32 + half * 8 + 2 * v + (v >= 4 ? 8 : 0);          // < 128 always
        int col = nt * 16 + n;
        sW1[i] = packbf(Wu1[k0 * 64 + col], Wu1[(k0 + 1) * 64 + col]);
    }
    for (int i = tid; i < 8 * 256; i += 256) {
        int tile = i >> 8;
        int lane = (i >> 3) & 31;
        int v    = i & 7;
        int kc = tile >> 2, nt = tile & 3;
        int half = lane >> 4, n = lane & 15;
        int k0 = kc * 32 + half * 8 + 2 * v + (v >= 4 ? 8 : 0);          // < 64 always
        int col = nt * 16 + n;
        sW2[i] = packbf(Wu2[k0 * 64 + col], Wu2[(k0 + 1) * 64 + col]);
    }
    __syncthreads();

    const int lane = tid & 31;
    const int wave = tid >> 5;
    const int half = lane >> 4;
    const int m    = lane & 15;
    const int n16  = lane & 15;
    float* hid = &sHid[wave][0];

    float bias1[4], bias2[4];
#pragma unroll
    for (int nt = 0; nt < 4; ++nt) {
        bias1[nt] = bu1[nt * 16 + n16];
        bias2[nt] = bu2[nt * 16 + n16];
    }

    const int gwave  = blockIdx.x * 8 + wave;
    const int nwaves = gridDim.x * 8;
    const int o = half * 8;

    for (int t = gwave; t < ntiles; t += nwaves) {
        const int n0  = t * 16;
        const int nid = min(n0 + m, N - 1);
        const float* hn = nodef + (long)nid * 64;
        const float* ha = agg   + (long)nid * 64;

        v16bf A0 = makeA(*(const float4*)(hn + o),      *(const float4*)(hn + o + 4),
                         *(const float4*)(hn + 16 + o), *(const float4*)(hn + 20 + o));
        v16bf A1 = makeA(*(const float4*)(hn + 32 + o), *(const float4*)(hn + 36 + o),
                         *(const float4*)(hn + 48 + o), *(const float4*)(hn + 52 + o));
        v16bf A2 = makeA(*(const float4*)(ha + o),      *(const float4*)(ha + o + 4),
                         *(const float4*)(ha + 16 + o), *(const float4*)(ha + 20 + o));
        v16bf A3 = makeA(*(const float4*)(ha + 32 + o), *(const float4*)(ha + 36 + o),
                         *(const float4*)(ha + 48 + o), *(const float4*)(ha + 52 + o));

        // ---- layer 1 ----
#pragma unroll
        for (int nt = 0; nt < 4; ++nt) {
            float b = bias1[nt];
            v8f c = {b, b, b, b, b, b, b, b};
            c = wmma_bf16(A0, loadB(&sW1[(0 * 4 + nt) * 256], lane), c);
            c = wmma_bf16(A1, loadB(&sW1[(1 * 4 + nt) * 256], lane), c);
            c = wmma_bf16(A2, loadB(&sW1[(2 * 4 + nt) * 256], lane), c);
            c = wmma_bf16(A3, loadB(&sW1[(3 * 4 + nt) * 256], lane), c);
#pragma unroll
            for (int r = 0; r < 8; ++r)
                hid[(r + 8 * half) * HID_STRIDE + nt * 16 + n16] = fmaxf(c[r], 0.f);
        }

        v16bf H0, H1;
#pragma unroll
        for (int kc = 0; kc < 2; ++kc) {
            const float* hrow = hid + m * HID_STRIDE + kc * 32;
            v16bf a = makeA(*(const float4*)(hrow + o),      *(const float4*)(hrow + o + 4),
                            *(const float4*)(hrow + 16 + o), *(const float4*)(hrow + 20 + o));
            if (kc == 0) H0 = a; else H1 = a;
        }

        // ---- layer 2: write "updated" back into LDS staging ----
#pragma unroll
        for (int nt = 0; nt < 4; ++nt) {
            float b = bias2[nt];
            v8f c = {b, b, b, b, b, b, b, b};
            c = wmma_bf16(H0, loadB(&sW2[(0 * 4 + nt) * 256], lane), c);
            c = wmma_bf16(H1, loadB(&sW2[(1 * 4 + nt) * 256], lane), c);
#pragma unroll
            for (int r = 0; r < 8; ++r)
                hid[(r + 8 * half) * HID_STRIDE + nt * 16 + n16] = c[r];
        }

        // ---- residual + LayerNorm: lane pair (L, L^16) co-owns row m ----
        if (n0 + m < N) {
            const float* xrow = nodef + (long)(n0 + m) * 64;
            const float* urow = hid + m * HID_STRIDE;
            float s = 0.f, s2 = 0.f;
#pragma unroll
            for (int j = 0; j < 32; ++j) {
                float x = xrow[half * 32 + j] + urow[half * 32 + j];
                s += x; s2 += x * x;
            }
            s  += __shfl_xor(s, 16, 32);
            s2 += __shfl_xor(s2, 16, 32);
            float mu  = s * (1.f / 64.f);
            float var = s2 * (1.f / 64.f) - mu * mu;
            float inv = rsqrtf(var + 1e-5f);
            float* orow = out + (long)(n0 + m) * 64;
#pragma unroll
            for (int j = 0; j < 32; ++j) {
                int f = half * 32 + j;
                float x = xrow[f] + urow[f];
                orow[f] = (x - mu) * inv * gamma[f] + beta[f];
            }
        }
    }
}

// ---------------------------------------------------------------------------
// Host entry
// ---------------------------------------------------------------------------
extern "C" void kernel_launch(void* const* d_in, const int* in_sizes, int n_in,
                              void* d_out, int out_size, void* d_ws, size_t ws_size,
                              hipStream_t stream) {
    const float*      nodef = (const float*)d_in[0];
    const long long*  eidx  = (const long long*)d_in[1];   // int64 per reference
    const float*      efeat = (const float*)d_in[2];
    const float*      Wm1   = (const float*)d_in[3];
    const float*      bm1   = (const float*)d_in[4];
    const float*      Wm2   = (const float*)d_in[5];
    const float*      bm2   = (const float*)d_in[6];
    const float*      Wu1   = (const float*)d_in[7];
    const float*      bu1   = (const float*)d_in[8];
    const float*      Wu2   = (const float*)d_in[9];
    const float*      bu2   = (const float*)d_in[10];
    const float*      gamma = (const float*)d_in[11];
    const float*      beta  = (const float*)d_in[12];
    float*            out   = (float*)d_out;

    const int N = in_sizes[0] / 64;
    const int E = in_sizes[1] / 2;

    // Workspace layout: [ agg: N*64 f32 ][ nodefbf: N*64 bf16 ]
    float*  agg     = (float*)d_ws;
    size_t  aggB    = (size_t)N * 64 * sizeof(float);
    size_t  needBf  = aggB + (size_t)N * 64 * sizeof(unsigned short);
    const bool bf16Gather = (ws_size >= needBf);
    unsigned short* nodefbf = (unsigned short*)((char*)d_ws + aggB);

    // 1) zero the segment-sum accumulator (deterministic every call)
    const int n4 = N * 16;                                 // N*64/4 float4s
    int zblocks = (n4 + 255) / 256; if (zblocks > 4096) zblocks = 4096;
    zero_f4_kernel<<<zblocks, 256, 0, stream>>>((float4*)agg, n4);

    // 2) pre-convert node features to bf16 (gathered ~32x each in kernel 3)
    if (bf16Gather) {
        const int npairs = N * 32;
        int cblocks = (npairs + 255) / 256; if (cblocks > 2048) cblocks = 2048;
        f32_to_bf16_kernel<<<cblocks, 256, 0, stream>>>(
            nodef, (unsigned*)nodefbf, npairs);
    }

    // 3) edge MLP + scatter
    const int etiles = (E + 15) / 16;
    int eblocks = (etiles + 7) / 8;
    if (eblocks > 2048) eblocks = 2048;
    if (bf16Gather) {
        edge_mlp_kernel<true><<<eblocks, 256, 0, stream>>>(
            nodef, nodefbf, eidx, efeat, Wm1, bm1, Wm2, bm2, agg, E, etiles);
    } else {
        edge_mlp_kernel<false><<<eblocks, 256, 0, stream>>>(
            nodef, nodefbf, eidx, efeat, Wm1, bm1, Wm2, bm2, agg, E, etiles);
    }

    // 4) node MLP + residual + LayerNorm
    const int vtiles = (N + 15) / 16;
    int vblocks = (vtiles + 7) / 8;
    if (vblocks > 2048) vblocks = 2048;
    node_update_kernel<<<vblocks, 256, 0, stream>>>(
        nodef, agg, Wu1, bu1, Wu2, bu2, gamma, beta, out, N, vtiles);
}
